// CoPE_62654982914419
// MI455X (gfx1250) — compile-verified, matrix operand order checked by reading, not run on hardware
//
#include <hip/hip_runtime.h>

// CoPE fused kernel for MI455X (gfx1250, wave32).
// Shapes: b=4, h=12, l=1024, d=64, npos=64.  M = b*h*l = 49152 rows.
//
// Memory-bound: ~415 MB of traffic -> ~18us floor @ 23.3 TB/s. The
// query@pos_emb GEMM (49152x64x64) is fused in via v_wmma_f32_16x16x32_bf16.

typedef __attribute__((ext_vector_type(16))) __bf16 bf16x16;
typedef __attribute__((ext_vector_type(8)))  float  f32x8;
typedef __attribute__((ext_vector_type(4)))  float  f32x4;   // native vector: OK for NT builtins

#define SEQ_L    1024
#define DHEAD    64
#define NPOS     64
#define ROWS_PB  16      // rows per block = one WMMA M-tile
#define NTHREADS 512     // 16 wave32s

__device__ __forceinline__ float sigmoidf_fast(float x) {
  return 1.0f / (1.0f + __expf(-x));
}

// clamp, floor/ceil gather from this row's 64 position logits, lerp
__device__ __forceinline__ float interp1(float p, const float* __restrict__ lrow) {
  p = fminf(p, 63.0f);
  float pf = floorf(p);
  int   fi = (int)pf;
  int   ci = (int)ceilf(p);
  float w  = p - pf;
  float lf = lrow[fi];
  float lc = lrow[ci];
  return fmaf(w, lc - lf, lf);
}

__global__ __launch_bounds__(NTHREADS) void cope_fused_kernel(
    const float* __restrict__ query,   // [M, 64]
    const float* __restrict__ attn,    // [M, 1024]
    const float* __restrict__ pe,      // [64, 64]  (d-major, row stride = NPOS)
    float* __restrict__ out)           // [M, 1024]
{
  __shared__ float s_pe[DHEAD * NPOS];        // 16 KB staged pos_emb
  __shared__ float s_q [ROWS_PB * DHEAD];     //  4 KB staged query rows
  __shared__ float s_lg[ROWS_PB * NPOS];      //  4 KB logits_int tile (16x64)

  const int tid  = threadIdx.x;
  const int lane = tid & 31;
  const int wave = tid >> 5;
  const long long rowBase = (long long)blockIdx.x * ROWS_PB;

  // ---- stage inputs to LDS (coalesced) ----
  for (int i = tid; i < DHEAD * NPOS; i += NTHREADS) s_pe[i] = pe[i];
  for (int i = tid; i < ROWS_PB * DHEAD; i += NTHREADS)
    s_q[i] = query[rowBase * DHEAD + i];
  __syncthreads();

  // ---- WMMA phase: waves 0..3 each produce one 16x16 N-tile of logits_int ----
  // A: 16x32 bf16 per k-chunk, ISA layout: lanes 0-15 hold K{0..7,16..23},
  // lanes 16-31 hold K{8..15,24..31}; VGPR v<4 -> K=2v(+1), v>=4 -> K=16+2(v-4).
  if (wave < 4) {
    const int mn    = lane & 15;               // row (A) / col (B) index
    const int kbase = (lane < 16) ? 0 : 8;
    const int ncol  = wave * 16 + mn;          // global N column of this tile
    f32x8 acc = {};
    #pragma unroll
    for (int kc = 0; kc < 2; ++kc) {
      const int koff = kc * 32 + kbase;
      bf16x16 a, b;
      #pragma unroll
      for (int v = 0; v < 8; ++v) {
        const int kk = (v < 4) ? (2 * v) : (16 + 2 * (v - 4));
        const int k0 = koff + kk;
        a[2 * v]     = (__bf16)s_q[mn * DHEAD + k0];
        a[2 * v + 1] = (__bf16)s_q[mn * DHEAD + k0 + 1];
        b[2 * v]     = (__bf16)s_pe[(k0)     * NPOS + ncol];
        b[2 * v + 1] = (__bf16)s_pe[(k0 + 1) * NPOS + ncol];
      }
      acc = __builtin_amdgcn_wmma_f32_16x16x32_bf16(
          /*neg_a=*/false, a, /*neg_b=*/false, b,
          /*c_mod=*/(short)0, acc, /*reuse_a=*/false, /*reuse_b=*/false);
    }
    // C/D layout: VGPR r -> M = r (lanes 0-15) or r+8 (lanes 16-31), N = lane&15
    const int mrow = (lane < 16) ? 0 : 8;
    #pragma unroll
    for (int r = 0; r < 8; ++r)
      s_lg[(mrow + r) * NPOS + ncol] = acc[r];
  }
  __syncthreads();

  // ---- main phase: one wave per row; reverse scan over 8 chunks of 128 ----
  const long long rowg = rowBase + wave;
  const float* __restrict__ arow = attn + rowg * SEQ_L;
  float*       __restrict__ orow = out  + rowg * SEQ_L;
  const float* __restrict__ lrow = &s_lg[wave * NPOS];
  const int diag = (int)(rowg & (SEQ_L - 1));   // i index within [l,l] tile

  float carry = 0.0f;                            // sum of gates in later chunks
  for (int c = (SEQ_L / 128) - 1; c >= 0; --c) {
    const int base = c * 128 + lane * 4;
    f32x4 g4 = __builtin_nontemporal_load((const f32x4*)(arow + base));

    float g0 = (base + 0 == diag) ? 0.0f : sigmoidf_fast(g4.x);
    float g1 = (base + 1 == diag) ? 0.0f : sigmoidf_fast(g4.y);
    float g2 = (base + 2 == diag) ? 0.0f : sigmoidf_fast(g4.z);
    float g3 = (base + 3 == diag) ? 0.0f : sigmoidf_fast(g4.w);

    // in-lane inclusive suffix sums (element k sums elements >= k)
    float s3 = g3;
    float s2 = g2 + s3;
    float s1 = g1 + s2;
    float s0 = g0 + s1;

    // cross-lane inclusive suffix scan of per-lane totals (Hillis-Steele)
    float S = s0;
    #pragma unroll
    for (int off = 1; off < 32; off <<= 1) {
      int   src = lane + off;
      float o   = __shfl(S, src & 31);
      S += (src < 32) ? o : 0.0f;
    }
    float E     = S - s0;          // exclusive suffix: lanes strictly right
    float total = __shfl(S, 0);    // whole-chunk gate sum

    f32x4 o4;
    o4.x = interp1(s0 + E + carry, lrow);
    o4.y = interp1(s1 + E + carry, lrow);
    o4.z = interp1(s2 + E + carry, lrow);
    o4.w = interp1(s3 + E + carry, lrow);
    __builtin_nontemporal_store(o4, (f32x4*)(orow + base));

    carry += total;
  }
}

extern "C" void kernel_launch(void* const* d_in, const int* in_sizes, int n_in,
                              void* d_out, int out_size, void* d_ws, size_t ws_size,
                              hipStream_t stream) {
  const float* query = (const float*)d_in[0];   // [b,h,l,d] f32
  const float* attn  = (const float*)d_in[1];   // [b,h,l,l] f32
  const float* pe    = (const float*)d_in[2];   // [d,npos]  f32
  float* out = (float*)d_out;                    // [b,h,l,l] f32

  const int M      = in_sizes[0] / DHEAD;        // b*h*l = 49152
  const int blocks = M / ROWS_PB;                // 3072
  cope_fused_kernel<<<blocks, NTHREADS, 0, stream>>>(query, attn, pe, out);
}
